// TransformerBlockWithMoE_43688407335203
// MI455X (gfx1250) — matrix operand order repaired and checked by reading, not compiled, not run on hardware
//
#include <hip/hip_runtime.h>

// ---------------------------------------------------------------------------
// CDNA5 (gfx1250) wave32 WMMA transformer block + hierarchical top-1 MoE.
// GEMMs: v_wmma_f32_16x16x32_bf16; LDS tiles staged in fragment-major order
// so every WMMA operand is two ds_load_b128. fp32->bf16 via native bf16 cvt.
// ---------------------------------------------------------------------------

typedef __attribute__((ext_vector_type(16))) __bf16          v16bf;
typedef __attribute__((ext_vector_type(2)))  __bf16          v2bf;
typedef __attribute__((ext_vector_type(8)))  float           v8f;

struct U32x8 { uint4 a, b; };

// fp32 -> bf16 (RNE). Prefer the packed cvt builtin; else the C-level cast
// (clang fptrunc -> native v_cvt_*bf16* on gfx1250).
__device__ __forceinline__ unsigned short bf16_bits(float f) {
  return __builtin_bit_cast(unsigned short, (__bf16)f);
}
__device__ __forceinline__ unsigned pack2(float a, float b) {
#if __has_builtin(__builtin_amdgcn_cvt_pk_bf16_f32)
  const v2bf p = __builtin_amdgcn_cvt_pk_bf16_f32(a, b);
  return __builtin_bit_cast(unsigned, p);
#else
  const v2bf p = { (__bf16)a, (__bf16)b };
  return __builtin_bit_cast(unsigned, p);
#endif
}

__device__ __forceinline__ v8f wmma_bf16(v16bf a, v16bf b, v8f c) {
  return __builtin_amdgcn_wmma_f32_16x16x32_bf16(
      false, a, false, b, (short)0, c, false, false);
}

// Fragment = 16 contiguous bf16 in LDS -> two 128-bit ds loads.
__device__ __forceinline__ v16bf load_frag16(const unsigned short* p) {
  U32x8 u;
  u.a = *(const uint4*)(p);
  u.b = *(const uint4*)(p + 8);
  return __builtin_bit_cast(v16bf, u);
}

// A-fragment swizzle (16x32 bf16, this code's contiguous k order, not the ISA table order): window-local K index kk ->
//   laneHalf = kk>>4  (lanes 0-15 vs 16-31), elem = kk&15
// Storage: Asw[(mblock*32 + mrow + laneHalf*16)*24 + elem], stride 24 ushorts.
#define A_LDS 24
#define B_LDS 40

// ---------------------------------------------------------------------------
// C[M,N] = A[M,K] @ W[N,K]^T + bias.  Tile 128x64, K-step 32.
// 8 waves: wave(wid) -> rows [32*(wid&3)), cols [32*(wid>>2)) : 4 WMMA/step.
// ---------------------------------------------------------------------------
__global__ __launch_bounds__(256)
void gemm_xwT_bias(const float* __restrict__ A, const float* __restrict__ W,
                   const float* __restrict__ bias, float* __restrict__ C,
                   int M, int N, int K) {
  __shared__ __align__(16) unsigned short Asw[8 * 32 * A_LDS];
  __shared__ __align__(16) unsigned short Btw[64 * B_LDS];
  const int tid = threadIdx.x, lane = tid & 31, wid = tid >> 5;
  const int mq = wid & 3, nh = wid >> 2;
  const int m0 = blockIdx.y * 128, n0 = blockIdx.x * 64;
  v8f c00{}, c01{}, c10{}, c11{};
  for (int k0 = 0; k0 < K; k0 += 32) {
    if (k0 + 32 < K) {
      __builtin_prefetch(A + (size_t)(m0 + (tid >> 1)) * K + k0 + 32, 0, 1);
      __builtin_prefetch(W + (size_t)(n0 + (tid >> 2)) * K + k0 + 32, 0, 1);
    }
    __syncthreads();
#pragma unroll
    for (int i = 0; i < 4; ++i) {            // A tile 128x32 -> swizzled bf16
      const int id = tid + i * 256;
      const int r = id >> 3, q = id & 7;     // row, K-quad (kk = 4q)
      const float4 v = *(const float4*)(A + (size_t)(m0 + r) * K + k0 + 4 * q);
      const int lh = q >> 2;
      const int eb = 4 * (q & 3);
      unsigned short* d = Asw + ((r >> 4) * 32 + (r & 15) + lh * 16) * A_LDS + eb;
      *(uint2*)d = make_uint2(pack2(v.x, v.y), pack2(v.z, v.w));
    }
#pragma unroll
    for (int i = 0; i < 2; ++i) {            // W[N,K] tile -> Btw[n][k]
      const int id = tid + i * 256;
      const int n = id >> 3, q = id & 7;
      const float4 v = *(const float4*)(W + (size_t)(n0 + n) * K + k0 + 4 * q);
      *(uint2*)(Btw + n * B_LDS + 4 * q) = make_uint2(pack2(v.x, v.y), pack2(v.z, v.w));
    }
    __syncthreads();
    const v16bf a0 = load_frag16(Asw + ((2 * mq + 0) * 32 + lane) * A_LDS);
    const v16bf a1 = load_frag16(Asw + ((2 * mq + 1) * 32 + lane) * A_LDS);
    const unsigned short* bp = Btw + (32 * nh + (lane & 15)) * B_LDS + (lane & 16);
    const v16bf b0 = load_frag16(bp);
    const v16bf b1 = load_frag16(bp + 16 * B_LDS);
    c00 = wmma_bf16(a0, b0, c00); c01 = wmma_bf16(a0, b1, c01);
    c10 = wmma_bf16(a1, b0, c10); c11 = wmma_bf16(a1, b1, c11);
  }
  const int colA = n0 + 32 * nh + (lane & 15);
  const int colB = colA + 16;
  const float bA = bias[colA], bB = bias[colB];
  const int rb = m0 + 32 * mq + ((lane & 16) ? 8 : 0);
#pragma unroll
  for (int r = 0; r < 8; ++r) {
    C[(size_t)(rb + r) * N + colA]      = c00[r] + bA;
    C[(size_t)(rb + r) * N + colB]      = c01[r] + bB;
    C[(size_t)(rb + 16 + r) * N + colA] = c10[r] + bA;
    C[(size_t)(rb + 16 + r) * N + colB] = c11[r] + bB;
  }
}

// ---------------------------------------------------------------------------
// Flash attention: block per (b*16+h, q-tile of 64). S=1024, hd=64.
// Q,P staged in A-fragment swizzle; K,V^T staged k-contiguous per column.
// Online softmax parallelized 4 threads/row; P stored as 2x uint4 per thread.
// ---------------------------------------------------------------------------
#define KV_LDS 72
__global__ __launch_bounds__(256)
void attn_kernel(const float* __restrict__ qkv, float* __restrict__ attnO) {
  const int bh = blockIdx.x, qt = blockIdx.y;
  const int b = bh >> 4, hh = bh & 15;
  const int tid = threadIdx.x, lane = tid & 31, wid = tid >> 5;
  const int mr = wid & 3, nc = wid >> 2;
  __shared__ __align__(16) unsigned short Qsw[4 * 2 * 32 * A_LDS];
  __shared__ __align__(16) unsigned short Psw[4 * 2 * 32 * A_LDS];
  __shared__ __align__(16) unsigned short Ks[64 * KV_LDS];   // [key][hd]
  __shared__ __align__(16) unsigned short Vt[64 * KV_LDS];   // [hd][key]
  __shared__ float Sf[64 * 65];
  __shared__ float pmax[4 * 64], psum[4 * 64];
  __shared__ float mrow[64], lrow[64], arow[64];
  const size_t rowQ = (size_t)(b * 1024 + qt * 64);
  // Q tile 64x64, pre-scaled by 1/sqrt(64), stored swizzled (2 K-windows).
#pragma unroll
  for (int i = 0; i < 4; ++i) {
    const int id = tid + i * 256;
    const int r = id >> 4, q16 = id & 15;        // col quad: c = 4*q16
    const float4 v = *(const float4*)(qkv + (rowQ + r) * 3072 + hh * 64 + 4 * q16);
    const int w = q16 >> 3, qq = q16 & 7;
    const int lh = qq >> 2;
    const int eb = 4 * (qq & 3);
    unsigned short* d = Qsw + (((r >> 4) * 2 + w) * 32 + (r & 15) + lh * 16) * A_LDS + eb;
    *(uint2*)d = make_uint2(pack2(v.x * 0.125f, v.y * 0.125f),
                            pack2(v.z * 0.125f, v.w * 0.125f));
  }
  if (tid < 64) { mrow[tid] = -1e30f; lrow[tid] = 0.f; }
  v8f o0 = {}, o1 = {};
  for (int j = 0; j < 16; ++j) {
    __syncthreads();
    const size_t rowK = (size_t)(b * 1024 + j * 64);
#pragma unroll
    for (int i = 0; i < 4; ++i) {
      const int id = tid + i * 256;
      const int r = id >> 4, q = id & 15;
      const float4 vk = *(const float4*)(qkv + (rowK + r) * 3072 + 1024 + hh * 64 + 4 * q);
      const float4 vv = *(const float4*)(qkv + (rowK + r) * 3072 + 2048 + hh * 64 + 4 * q);
      // K: row-major [key][hd] == Bt layout for S = Q K^T
      *(uint2*)(Ks + r * KV_LDS + 4 * q) = make_uint2(pack2(vk.x, vk.y), pack2(vk.z, vk.w));
      // V: transpose into [hd][key] == Bt layout for O = P V
      unsigned short* dv = Vt + (4 * q) * KV_LDS + r;
      dv[0] = bf16_bits(vv.x); dv[KV_LDS] = bf16_bits(vv.y);
      dv[2 * KV_LDS] = bf16_bits(vv.z); dv[3 * KV_LDS] = bf16_bits(vv.w);
    }
    __syncthreads();
    v8f s0 = {}, s1 = {};
#pragma unroll
    for (int ks = 0; ks < 64; ks += 32) {
      const v16bf a = load_frag16(Qsw + ((mr * 2 + (ks >> 5)) * 32 + lane) * A_LDS);
      const unsigned short* kp = Ks + (32 * nc + (lane & 15)) * KV_LDS + ks + (lane & 16);
      s0 = wmma_bf16(a, load_frag16(kp), s0);
      s1 = wmma_bf16(a, load_frag16(kp + 16 * KV_LDS), s1);
    }
    {
      const int c0 = 32 * nc + (lane & 15);
      const int rb = 16 * mr + ((lane & 16) ? 8 : 0);
#pragma unroll
      for (int r = 0; r < 8; ++r) {
        Sf[(rb + r) * 65 + c0]      = s0[r];
        Sf[(rb + r) * 65 + c0 + 16] = s1[r];
      }
    }
    __syncthreads();
    {                                   // phase A: partial max, 4 thr/row
      const int r = tid & 63, qtr = tid >> 6;
      const float* srow = Sf + r * 65 + 16 * qtr;
      float pm = -1e30f;
#pragma unroll
      for (int c = 0; c < 16; ++c) pm = fmaxf(pm, srow[c]);
      pmax[qtr * 64 + r] = pm;
    }
    __syncthreads();
    if (tid < 64) {                     // phase B: new running max + alpha
      const int r = tid;
      const float mnew = fmaxf(mrow[r],
          fmaxf(fmaxf(pmax[r], pmax[64 + r]), fmaxf(pmax[128 + r], pmax[192 + r])));
      arow[r] = __expf(mrow[r] - mnew);
      mrow[r] = mnew;
    }
    __syncthreads();
    {                                   // phase C: exp + swizzled P store
      const int r = tid & 63, qtr = tid >> 6;
      const float mnew = mrow[r];
      const float* srow = Sf + r * 65 + 16 * qtr;
      // cols 16*qtr..+15: window = qtr>>1, laneHalf = qtr&1,
      // elemBase = 8*(c16>>3) -> 8 contiguous elems per half = one uint4 store.
      unsigned short* dbase = Psw
          + (((r >> 4) * 2 + (qtr >> 1)) * 32 + (r & 15) + (qtr & 1) * 16) * A_LDS;
      float sum = 0.f;
#pragma unroll
      for (int half = 0; half < 2; ++half) {
        const float p0 = __expf(srow[half * 8 + 0] - mnew);
        const float p1 = __expf(srow[half * 8 + 1] - mnew);
        const float p2 = __expf(srow[half * 8 + 2] - mnew);
        const float p3 = __expf(srow[half * 8 + 3] - mnew);
        const float p4 = __expf(srow[half * 8 + 4] - mnew);
        const float p5 = __expf(srow[half * 8 + 5] - mnew);
        const float p6 = __expf(srow[half * 8 + 6] - mnew);
        const float p7 = __expf(srow[half * 8 + 7] - mnew);
        sum += p0 + p1 + p2 + p3 + p4 + p5 + p6 + p7;
        *(uint4*)(dbase + half * 8) =
            make_uint4(pack2(p0, p1), pack2(p2, p3), pack2(p4, p5), pack2(p6, p7));
      }
      psum[qtr * 64 + r] = sum;
    }
    __syncthreads();
    if (tid < 64)                       // phase D: running denominator
      lrow[tid] = lrow[tid] * arow[tid]
                + psum[tid] + psum[64 + tid] + psum[128 + tid] + psum[192 + tid];
    {
      const int rb = 16 * mr + ((lane & 16) ? 8 : 0);
#pragma unroll
      for (int r = 0; r < 8; ++r) { const float al = arow[rb + r]; o0[r] *= al; o1[r] *= al; }
#pragma unroll
      for (int ks = 0; ks < 64; ks += 32) {
        const v16bf a = load_frag16(Psw + ((mr * 2 + (ks >> 5)) * 32 + lane) * A_LDS);
        const unsigned short* vp = Vt + (32 * nc + (lane & 15)) * KV_LDS + ks + (lane & 16);
        o0 = wmma_bf16(a, load_frag16(vp), o0);
        o1 = wmma_bf16(a, load_frag16(vp + 16 * KV_LDS), o1);
      }
    }
  }
  __syncthreads();                      // lrow final before epilogue reads
  const int rb = 16 * mr + ((lane & 16) ? 8 : 0);
  const int c0 = hh * 64 + 32 * nc + (lane & 15);
#pragma unroll
  for (int r = 0; r < 8; ++r) {
    const float inv = 1.0f / lrow[rb + r];
    const size_t row = rowQ + rb + r;
    attnO[row * 1024 + c0]      = o0[r] * inv;
    attnO[row * 1024 + c0 + 16] = o1[r] * inv;
  }
}

// ---------------------------------------------------------------------------
// out[row] = LayerNorm(a[row] + b[row]) * g + beta     (D = 1024)
// ---------------------------------------------------------------------------
__global__ __launch_bounds__(256)
void add_ln(const float* __restrict__ a, const float* __restrict__ bb,
            const float* __restrict__ g, const float* __restrict__ be,
            float* __restrict__ o) {
  const int row = blockIdx.x, tid = threadIdx.x;
  __shared__ float s1[256], s2[256];
  float v[4]; float sum = 0.f, sq = 0.f;
#pragma unroll
  for (int i = 0; i < 4; ++i) {
    const int c = tid + 256 * i;
    v[i] = a[(size_t)row * 1024 + c] + bb[(size_t)row * 1024 + c];
    sum += v[i]; sq += v[i] * v[i];
  }
  s1[tid] = sum; s2[tid] = sq;
  __syncthreads();
  for (int s = 128; s > 0; s >>= 1) {
    if (tid < s) { s1[tid] += s1[tid + s]; s2[tid] += s2[tid + s]; }
    __syncthreads();
  }
  const float mu = s1[0] * (1.0f / 1024.0f);
  const float var = s2[0] * (1.0f / 1024.0f) - mu * mu;
  const float rs = rsqrtf(var + 1e-5f);
#pragma unroll
  for (int i = 0; i < 4; ++i) {
    const int c = tid + 256 * i;
    o[(size_t)row * 1024 + c] = (v[i] - mu) * rs * g[c] + be[c];
  }
}

__global__ void init_counts(int* cnt) { if (threadIdx.x < 8) cnt[threadIdx.x] = 0; }

// ---------------------------------------------------------------------------
// Hierarchical top-1 routing: group gate (D->4) then expert gate (D->2).
// ---------------------------------------------------------------------------
__global__ __launch_bounds__(128)
void routing_kernel(const float* __restrict__ x1, const float* __restrict__ wg,
                    const float* __restrict__ bg, const float* __restrict__ we,
                    const float* __restrict__ be, int* __restrict__ cnt,
                    int* __restrict__ elist, float* __restrict__ prob) {
  const int t = blockIdx.x, tid = threadIdx.x;
  const float* xr = x1 + (size_t)t * 1024;
  __shared__ float red[4][128];
  __shared__ float red2[2][128];
  __shared__ int gsel;
  float acc[4] = {0.f, 0.f, 0.f, 0.f};
  for (int d = tid; d < 1024; d += 128) {
    const float xv = xr[d];
#pragma unroll
    for (int g = 0; g < 4; ++g) acc[g] += xv * wg[d * 4 + g];
  }
#pragma unroll
  for (int g = 0; g < 4; ++g) red[g][tid] = acc[g];
  __syncthreads();
  for (int s = 64; s > 0; s >>= 1) {
    if (tid < s)
#pragma unroll
      for (int g = 0; g < 4; ++g) red[g][tid] += red[g][tid + s];
    __syncthreads();
  }
  if (tid == 0) {
    float lg[4]; float mx = -1e30f; int gi = 0;
#pragma unroll
    for (int g = 0; g < 4; ++g) {
      lg[g] = red[g][0] + bg[g];
      if (lg[g] > mx) { mx = lg[g]; gi = g; }
    }
    float den = 0.f;
#pragma unroll
    for (int g = 0; g < 4; ++g) den += __expf(lg[g] - mx);
    red[0][0] = 1.0f / den;      // softmax prob at argmax
    gsel = gi;
  }
  __syncthreads();
  const int g = gsel;
  float e0 = 0.f, e1 = 0.f;
  for (int d = tid; d < 1024; d += 128) {
    const float xv = xr[d];
    e0 += xv * we[((size_t)g * 1024 + d) * 2 + 0];
    e1 += xv * we[((size_t)g * 1024 + d) * 2 + 1];
  }
  red2[0][tid] = e0; red2[1][tid] = e1;
  __syncthreads();
  for (int s = 64; s > 0; s >>= 1) {
    if (tid < s) { red2[0][tid] += red2[0][tid + s]; red2[1][tid] += red2[1][tid + s]; }
    __syncthreads();
  }
  if (tid == 0) {
    const float l0 = red2[0][0] + be[g * 2 + 0];
    const float l1 = red2[1][0] + be[g * 2 + 1];
    const int ei = (l1 > l0) ? 1 : 0;
    const float mx = fmaxf(l0, l1);
    const float ep = 1.0f / (__expf(l0 - mx) + __expf(l1 - mx));
    const int fe = g * 2 + ei;
    const int pos = atomicAdd(&cnt[fe], 1);
    elist[fe * 2048 + pos] = t;
    prob[t] = red[0][0] * ep;
  }
}

// ---------------------------------------------------------------------------
// MoE FC1: h[tok] = relu(x1[tok] @ w1[e] + b1[e]) as bf16. Gathered rows.
// grid: (2048/64=32, 2048/128=16, E=8); tile 128x64.
// ---------------------------------------------------------------------------
__global__ __launch_bounds__(256)
void moe_fc1(const float* __restrict__ x1, const float* __restrict__ w1,
             const float* __restrict__ b1, const int* __restrict__ cnt,
             const int* __restrict__ elist, unsigned short* __restrict__ h) {
  const int e = blockIdx.z, mt = blockIdx.y, nt = blockIdx.x;
  const int count = cnt[e];
  const int m0 = mt * 128;
  if (m0 >= count) return;
  __shared__ __align__(16) unsigned short Asw[8 * 32 * A_LDS];
  __shared__ __align__(16) unsigned short Btw[64 * B_LDS];
  __shared__ int rows[128];
  const int tid = threadIdx.x, lane = tid & 31, wid = tid >> 5;
  const int mq = wid & 3, nh = wid >> 2;
  if (tid < 128) rows[tid] = (m0 + tid < count) ? elist[e * 2048 + m0 + tid] : -1;
  const int n0 = nt * 64;
  const float* W = w1 + (size_t)e * 1024 * 2048;
  v8f c00{}, c01{}, c10{}, c11{};
  for (int k0 = 0; k0 < 1024; k0 += 32) {
    if (k0 + 32 < 1024)
      __builtin_prefetch(W + (size_t)(k0 + 32 + (tid >> 3)) * 2048 + n0, 0, 1);
    __syncthreads();
#pragma unroll
    for (int i = 0; i < 4; ++i) {            // gathered A tile 128x32
      const int id = tid + i * 256;
      const int r = id >> 3, q = id & 7;
      const int tok = rows[r];
      float4 v = {0.f, 0.f, 0.f, 0.f};
      if (tok >= 0) v = *(const float4*)(x1 + (size_t)tok * 1024 + k0 + 4 * q);
      const int lh = q >> 2;
      const int eb = 4 * (q & 3);
      unsigned short* d = Asw + ((r >> 4) * 32 + (r & 15) + lh * 16) * A_LDS + eb;
      *(uint2*)d = make_uint2(pack2(v.x, v.y), pack2(v.z, v.w));
    }
#pragma unroll
    for (int i = 0; i < 2; ++i) {            // B [K][N] row-major -> Btw[n][k]
      const int id = tid + i * 256;
      const int r = id >> 4, q = id & 15;    // r = k, q = n quad
      const float4 v = *(const float4*)(W + (size_t)(k0 + r) * 2048 + n0 + 4 * q);
      unsigned short* d = Btw + (4 * q) * B_LDS + r;
      d[0] = bf16_bits(v.x); d[B_LDS] = bf16_bits(v.y);
      d[2 * B_LDS] = bf16_bits(v.z); d[3 * B_LDS] = bf16_bits(v.w);
    }
    __syncthreads();
    const v16bf a0 = load_frag16(Asw + ((2 * mq + 0) * 32 + lane) * A_LDS);
    const v16bf a1 = load_frag16(Asw + ((2 * mq + 1) * 32 + lane) * A_LDS);
    const unsigned short* bp = Btw + (32 * nh + (lane & 15)) * B_LDS + (lane & 16);
    const v16bf b0 = load_frag16(bp);
    const v16bf b1 = load_frag16(bp + 16 * B_LDS);
    c00 = wmma_bf16(a0, b0, c00); c01 = wmma_bf16(a0, b1, c01);
    c10 = wmma_bf16(a1, b0, c10); c11 = wmma_bf16(a1, b1, c11);
  }
  const int colA = n0 + 32 * nh + (lane & 15);
  const int colB = colA + 16;
  const float* bias = b1 + (size_t)e * 2048;
  const float bA = bias[colA], bB = bias[colB];
  const int lo = (lane & 16) ? 8 : 0;
#pragma unroll
  for (int r = 0; r < 8; ++r) {
#pragma unroll
    for (int i = 0; i < 2; ++i) {
      const int rl = 32 * mq + 16 * i + lo + r;
      if (m0 + rl < count) {
        const int tok = rows[rl];
        const float vA = (i == 0) ? c00[r] : c10[r];
        const float vB = (i == 0) ? c01[r] : c11[r];
        h[(size_t)tok * 2048 + colA] = bf16_bits(fmaxf(vA + bA, 0.f));
        h[(size_t)tok * 2048 + colB] = bf16_bits(fmaxf(vB + bB, 0.f));
      }
    }
  }
}

// ---------------------------------------------------------------------------
// MoE FC2: out[tok] = prob[tok] * (h[tok] @ w2[e] + b2[e]). Scatter by token.
// grid: (1024/64=16, 16, E=8); tile 128x64; A is already bf16.
// ---------------------------------------------------------------------------
__global__ __launch_bounds__(256)
void moe_fc2(const unsigned short* __restrict__ h, const float* __restrict__ w2,
             const float* __restrict__ b2, const int* __restrict__ cnt,
             const int* __restrict__ elist, const float* __restrict__ prob,
             float* __restrict__ out) {
  const int e = blockIdx.z, mt = blockIdx.y, nt = blockIdx.x;
  const int count = cnt[e];
  const int m0 = mt * 128;
  if (m0 >= count) return;
  __shared__ __align__(16) unsigned short Asw[8 * 32 * A_LDS];
  __shared__ __align__(16) unsigned short Btw[64 * B_LDS];
  __shared__ int rows[128];
  const int tid = threadIdx.x, lane = tid & 31, wid = tid >> 5;
  const int mq = wid & 3, nh = wid >> 2;
  if (tid < 128) rows[tid] = (m0 + tid < count) ? elist[e * 2048 + m0 + tid] : -1;
  const int n0 = nt * 64;
  const float* W = w2 + (size_t)e * 2048 * 1024;
  v8f c00{}, c01{}, c10{}, c11{};
  for (int k0 = 0; k0 < 2048; k0 += 32) {
    if (k0 + 32 < 2048)
      __builtin_prefetch(W + (size_t)(k0 + 32 + (tid >> 3)) * 1024 + n0, 0, 1);
    __syncthreads();
#pragma unroll
    for (int i = 0; i < 2; ++i) {            // gathered bf16 A tile 128x32
      const int id = tid + i * 256;          // 512 chunks of 8 bf16
      const int r = id >> 2, c8 = id & 3;    // kk = 8*c8
      const int tok = rows[r];
      uint4 v = make_uint4(0, 0, 0, 0);
      if (tok >= 0) v = *(const uint4*)(h + (size_t)tok * 2048 + k0 + 8 * c8);
      // chunk c8: laneHalf = c8>>1, elems (c8&1 ? 8..15 : 0..7) contiguous
      unsigned short* d = Asw + ((r >> 4) * 32 + (r & 15) + (c8 >> 1) * 16) * A_LDS
                        + ((c8 & 1) ? 8 : 0);
      *(uint4*)d = v;
    }
#pragma unroll
    for (int i = 0; i < 2; ++i) {            // B [K][N] row-major -> Btw[n][k]
      const int id = tid + i * 256;
      const int r = id >> 4, q = id & 15;
      const float4 v = *(const float4*)(W + (size_t)(k0 + r) * 1024 + n0 + 4 * q);
      unsigned short* d = Btw + (4 * q) * B_LDS + r;
      d[0] = bf16_bits(v.x); d[B_LDS] = bf16_bits(v.y);
      d[2 * B_LDS] = bf16_bits(v.z); d[3 * B_LDS] = bf16_bits(v.w);
    }
    __syncthreads();
    const v16bf a0 = load_frag16(Asw + ((2 * mq + 0) * 32 + lane) * A_LDS);
    const v16bf a1 = load_frag16(Asw + ((2 * mq + 1) * 32 + lane) * A_LDS);
    const unsigned short* bp = Btw + (32 * nh + (lane & 15)) * B_LDS + (lane & 16);
    const v16bf b0 = load_frag16(bp);
    const v16bf b1 = load_frag16(bp + 16 * B_LDS);
    c00 = wmma_bf16(a0, b0, c00); c01 = wmma_bf16(a0, b1, c01);
    c10 = wmma_bf16(a1, b0, c10); c11 = wmma_bf16(a1, b1, c11);
  }
  const int colA = n0 + 32 * nh + (lane & 15);
  const int colB = colA + 16;
  const float* bias = b2 + (size_t)e * 1024;
  const float bA = bias[colA], bB = bias[colB];
  const int lo = (lane & 16) ? 8 : 0;
#pragma unroll
  for (int r = 0; r < 8; ++r) {
#pragma unroll
    for (int i = 0; i < 2; ++i) {
      const int rl = 32 * mq + 16 * i + lo + r;
      if (m0 + rl < count) {
        const int tok = rows[rl];
        const float p = prob[tok];
        const float vA = (i == 0) ? c00[r] : c10[r];
        const float vB = (i == 0) ? c01[r] : c11[r];
        out[(size_t)tok * 1024 + colA] = p * (vA + bA);
        out[(size_t)tok * 1024 + colB] = p * (vB + bB);
      }
    }
  }
}

// ---------------------------------------------------------------------------
extern "C" void kernel_launch(void* const* d_in, const int* in_sizes, int n_in,
                              void* d_out, int out_size, void* d_ws, size_t ws_size,
                              hipStream_t stream) {
  const float* x    = (const float*)d_in[0];
  const float* wi   = (const float*)d_in[1];
  const float* bi   = (const float*)d_in[2];
  const float* wo   = (const float*)d_in[3];
  const float* bo   = (const float*)d_in[4];
  const float* g1   = (const float*)d_in[5];
  const float* be1  = (const float*)d_in[6];
  const float* g2   = (const float*)d_in[7];
  const float* be2  = (const float*)d_in[8];
  const float* wg   = (const float*)d_in[9];
  const float* bg   = (const float*)d_in[10];
  const float* we   = (const float*)d_in[11];
  const float* bexp = (const float*)d_in[12];
  const float* w1   = (const float*)d_in[13];
  const float* b1   = (const float*)d_in[14];
  const float* w2   = (const float*)d_in[15];
  const float* b2   = (const float*)d_in[16];

  float* ws    = (float*)d_ws;
  float* qkv   = ws;                       // 2048*3072
  float* attnO = qkv   + 2048ull * 3072;   // 2048*1024
  float* attnP = attnO + 2048ull * 1024;   // 2048*1024
  float* x1    = attnP + 2048ull * 1024;   // 2048*1024
  float* moeo  = x1    + 2048ull * 1024;   // 2048*1024
  float* prob  = moeo  + 2048ull * 1024;   // 2048
  int*   cnt   = (int*)(prob + 2048);      // 8
  int*   elist = cnt + 8;                  // 8*2048
  unsigned short* h = (unsigned short*)(elist + 8 * 2048);  // 2048*2048 bf16

  const dim3 blk(256);
  gemm_xwT_bias<<<dim3(48, 16), blk, 0, stream>>>(x, wi, bi, qkv, 2048, 3072, 1024);
  attn_kernel<<<dim3(32, 16), blk, 0, stream>>>(qkv, attnO);
  gemm_xwT_bias<<<dim3(16, 16), blk, 0, stream>>>(attnO, wo, bo, attnP, 2048, 1024, 1024);
  add_ln<<<2048, 256, 0, stream>>>(x, attnP, g1, be1, x1);
  init_counts<<<1, 32, 0, stream>>>(cnt);
  routing_kernel<<<2048, 128, 0, stream>>>(x1, wg, bg, we, bexp, cnt, elist, prob);
  moe_fc1<<<dim3(32, 16, 8), blk, 0, stream>>>(x1, w1, b1, cnt, elist, h);
  moe_fc2<<<dim3(16, 16, 8), blk, 0, stream>>>(h, w2, b2, cnt, elist, prob, moeo);
  add_ln<<<2048, 256, 0, stream>>>(x1, moeo, g2, be2, (float*)d_out);
}
